// LocalAttention_83236466196833
// MI455X (gfx1250) — matrix operand order repaired and checked by reading, not compiled, not run on hardware
//
#include <hip/hip_runtime.h>
#include <hip/hip_bf16.h>

// ---------------------------------------------------------------------------
// Local (3x3-neighborhood) attention, MI455X / gfx1250, wave32.
//   B=16, N=1024 (32x32 grid), C=1024, H=16, D=64.
// Pipeline:
//   1) f32->f16 converts (x, w_qkv, w_proj)
//   2) QKV GEMM  (16384x1024)x(1024x3072)  via v_wmma_f32_16x16x32_f16,
//      tiles staged with GLOBAL_LOAD_ASYNC_TO_LDS_B128 (ASYNCcnt), double-buffered
//   3) local attention (<=9 keys/query) in VALU, wave per (b,h,query)
//   4) Proj GEMM (16384x1024)x(1024x1024) + bias -> fp32 d_out
// Roofline: GEMMs ~137 GFLOP (tiny at WMMA rates); HBM bytes ~200MB -> ~10us
// at 23.3 TB/s; weights + f16 activations fit in the 192MB L2. Async LDS
// staging removes the VGPR round-trip and overlaps global traffic with WMMA.
// The K-loop is peeled so the steady state is branch-free:
//   stage(next) -> s_wait_asynccnt 4 -> barrier -> 8x WMMA -> barrier.
// ---------------------------------------------------------------------------

typedef _Float16 f16;
typedef __attribute__((ext_vector_type(16))) _Float16 v16h;
typedef __attribute__((ext_vector_type(8)))  _Float16 v8h;
typedef __attribute__((ext_vector_type(8)))  float    v8f;

#define DIMC   1024
#define HEADS  16
#define BATCH  16
#define NTOK   1024   // 32*32

// ---------------------------------------------------------------------------
// fp32 -> fp16 convert (grid-stride)
// ---------------------------------------------------------------------------
__global__ __launch_bounds__(256) void cvt_f32_f16(const float* __restrict__ in,
                                                   f16* __restrict__ out, size_t n) {
    size_t i = (size_t)blockIdx.x * blockDim.x + threadIdx.x;
    size_t stride = (size_t)gridDim.x * blockDim.x;
    for (; i < n; i += stride) out[i] = (f16)in[i];
}

// ---------------------------------------------------------------------------
// Tiled WMMA GEMM: C[M,N] = A[M,K] * Bw[N,K]^T   (A, Bw row-major f16)
// Block tile 128x128, BK=32, 256 threads = 8 waves (2x4), wave tile 64x32,
// 4x2 accumulators of v_wmma_f32_16x16x32_f16.
// Staging: global_load_async_to_lds_b128, double-buffered, s_wait_asynccnt.
// EPI==0: scatter to q/k/v f16 buffers, layout (B,H,N,D); n -> (which,h,d).
// EPI==1: out[m*N+n] = acc + bias[n] (fp32).
// ---------------------------------------------------------------------------
template <int EPI>
__global__ __launch_bounds__(256) void gemm_wmma_f16(
    const f16* __restrict__ A, const f16* __restrict__ Bw,
    int M, int N, int K,
    f16* __restrict__ qo, f16* __restrict__ ko, f16* __restrict__ vo,
    const float* __restrict__ bias, float* __restrict__ out) {

    // LDS: [bufA0 | bufA1 | bufB0 | bufB1], each 128 rows x 40 halfs
    // (32 data halfs + 8 pad; 80B row stride keeps 16B alignment, spreads banks)
    __shared__ f16 ldsbuf[4 * 128 * 40];
    const int TILEH = 128 * 40;                       // halfs per tile
    const unsigned TILEB = TILEH * 2;                 // bytes per tile
    const unsigned lds_base = (unsigned)(size_t)(void*)ldsbuf;  // LDS byte offset

    const int tid  = threadIdx.x;
    const int lane = tid & 31;
    const int wave = tid >> 5;
    const int wm   = wave >> 2;   // 0..1  (64 rows each)
    const int wn   = wave & 3;    // 0..3  (32 cols each)
    const int tile_n = blockIdx.x;
    const int tile_m = blockIdx.y;

    const int lr = lane & 15;     // row-within-16 / col-within-16
    const int hl = lane >> 4;     // lane half
    const int kbA = hl * 8;       // A frag: K base {0,8} (+{0,16} across vgpr halves)
    const int kbB = hl * 8;       // B frag: K base {0,8} (+{0,16} across vgpr halves)

    // Per-thread staging slots: thread covers (r0,col) and (r1,col), 8 halfs each,
    // for both A and B tiles (4 async b128 transfers per stage per thread).
    const int r0  = tid >> 2;            // 0..63
    const int r1  = r0 + 64;             // 64..127
    const int col = (tid & 3) << 3;      // 0,8,16,24

    const unsigned long long gA0 =
        (unsigned long long)(A + (size_t)(tile_m * 128 + r0) * K + col);
    const unsigned long long gA1 =
        (unsigned long long)(A + (size_t)(tile_m * 128 + r1) * K + col);
    const unsigned long long gB0 =
        (unsigned long long)(Bw + (size_t)(tile_n * 128 + r0) * K + col);
    const unsigned long long gB1 =
        (unsigned long long)(Bw + (size_t)(tile_n * 128 + r1) * K + col);

    const unsigned lA0 = (unsigned)(r0 * 40 + col) * 2;
    const unsigned lA1 = (unsigned)(r1 * 40 + col) * 2;

    // Issue one K-tile's 4 async global->LDS transfers (per thread).
    auto stage = [&](int k0, int buf) {
        unsigned long long kb = (unsigned long long)k0 * 2;   // byte offset in K
        unsigned a0 = lds_base + (unsigned)buf * TILEB + lA0;
        unsigned a1 = lds_base + (unsigned)buf * TILEB + lA1;
        unsigned b0 = lds_base + (unsigned)(2 + buf) * TILEB + lA0;
        unsigned b1 = lds_base + (unsigned)(2 + buf) * TILEB + lA1;
        asm volatile("global_load_async_to_lds_b128 %0, %1, off"
                     :: "v"(a0), "v"(gA0 + kb) : "memory");
        asm volatile("global_load_async_to_lds_b128 %0, %1, off"
                     :: "v"(a1), "v"(gA1 + kb) : "memory");
        asm volatile("global_load_async_to_lds_b128 %0, %1, off"
                     :: "v"(b0), "v"(gB0 + kb) : "memory");
        asm volatile("global_load_async_to_lds_b128 %0, %1, off"
                     :: "v"(b1), "v"(gB1 + kb) : "memory");
    };

    v8f acc[4][2];
#pragma unroll
    for (int tm = 0; tm < 4; ++tm)
#pragma unroll
        for (int tn = 0; tn < 2; ++tn) acc[tm][tn] = (v8f)(0.0f);

    // Fragment loads + 8 WMMAs on buffer `cur`.
    auto compute = [&](int cur) {
        const f16* Acur = &ldsbuf[cur * TILEH];
        const f16* Bcur = &ldsbuf[(2 + cur) * TILEH];

        v16h afrag[4], bfrag[2];
#pragma unroll
        for (int tm = 0; tm < 4; ++tm) {
            int row = wm * 64 + tm * 16 + lr;
            union { v16h v; v8h h[2]; } u;
            u.h[0] = *(const v8h*)(&Acur[row * 40 + kbA]);        // K kb..kb+7
            u.h[1] = *(const v8h*)(&Acur[row * 40 + 16 + kbA]);   // K kb+16..kb+23
            afrag[tm] = u.v;
        }
#pragma unroll
        for (int tn = 0; tn < 2; ++tn) {
            int row = wn * 32 + tn * 16 + lr;
            union { v16h v; v8h h[2]; } u;
            u.h[0] = *(const v8h*)(&Bcur[row * 40 + kbB]);        // K kb..kb+7
            u.h[1] = *(const v8h*)(&Bcur[row * 40 + 16 + kbB]);   // K kb+16..kb+23
            bfrag[tn] = u.v;
        }
#pragma unroll
        for (int tm = 0; tm < 4; ++tm)
#pragma unroll
            for (int tn = 0; tn < 2; ++tn)
                acc[tm][tn] = __builtin_amdgcn_wmma_f32_16x16x32_f16(
                    false, afrag[tm], false, bfrag[tn],
                    (short)0, acc[tm][tn], false, false);
    };

    const int KT = K / 32;
    stage(0, 0);   // prologue: tile 0 -> buffer 0

    // Steady state (KT-1 iterations, branch-free body).
    for (int kt = 0; kt < KT - 1; ++kt) {
        const int cur = kt & 1;
        stage((kt + 1) * 32, 1 - cur);     // prefetch next tile
        // 4 newest async ops (next tile) may remain outstanding;
        // everything older (current tile) is complete.
        asm volatile("s_wait_asynccnt 0x4" ::: "memory");
        __syncthreads();                   // current tile visible to all waves
        compute(cur);
        __syncthreads();                   // reads done before buf re-staged
    }

    // Peeled final tile.
    asm volatile("s_wait_asynccnt 0x0" ::: "memory");
    __syncthreads();
    compute((KT - 1) & 1);

    // Epilogue. C/D layout: col = lane%16, row = vgpr + 8*(lane>=16).
#pragma unroll
    for (int tm = 0; tm < 4; ++tm) {
#pragma unroll
        for (int tn = 0; tn < 2; ++tn) {
#pragma unroll
            for (int e = 0; e < 8; ++e) {
                int m = tile_m * 128 + wm * 64 + tm * 16 + e + hl * 8;
                int n = tile_n * 128 + wn * 32 + tn * 16 + lr;
                float val = acc[tm][tn][e];
                if (EPI == 0) {
                    int which = n >> 10;            // 0:q 1:k 2:v
                    int h     = (n & 1023) >> 6;
                    int d     = n & 63;
                    int b     = m >> 10;
                    int i     = m & 1023;
                    size_t dst = ((size_t)(b * HEADS + h) * NTOK + i) * 64 + d;
                    f16* p = (which == 0) ? qo : (which == 1) ? ko : vo;
                    p[dst] = (f16)val;
                } else {
                    out[(size_t)m * N + n] = val + bias[n];
                }
            }
        }
    }
}

// ---------------------------------------------------------------------------
// Local attention: one wave per (b,h,query). lane holds channels d=lane and
// d=lane+32. 9 neighbor scores via shfl_xor butterfly; stable softmax; V accum.
// Output yh in (B,N,C) layout with channel = h*64+d (feeds proj GEMM).
// ---------------------------------------------------------------------------
__global__ __launch_bounds__(256) void attn_local(
    const f16* __restrict__ qh, const f16* __restrict__ kh,
    const f16* __restrict__ vh, f16* __restrict__ yh) {

    const int wid  = blockIdx.x * 8 + (threadIdx.x >> 5);   // (b,h,i) flat
    const int lane = threadIdx.x & 31;
    const int b = wid >> 14;
    const int rem = wid & 16383;
    const int h = rem >> 10;
    const int i = rem & 1023;
    const int r = i >> 5, c = i & 31;

    const size_t base = (size_t)(b * HEADS + h) * NTOK * 64;

    const float scale = 0.125f;   // D=64 -> 1/sqrt(64)
    float q0 = (float)qh[base + (size_t)i * 64 + lane] * scale;
    float q1 = (float)qh[base + (size_t)i * 64 + lane + 32] * scale;

    float sc[9];
    int   jn[9];
#pragma unroll
    for (int t = 0; t < 9; ++t) {
        int dr = t / 3 - 1, dc = t % 3 - 1;
        int rr = r + dr, cc = c + dc;
        bool valid = (rr >= 0) & (rr < 32) & (cc >= 0) & (cc < 32);
        int j = valid ? (rr * 32 + cc) : i;
        jn[t] = j;
        float k0 = (float)kh[base + (size_t)j * 64 + lane];
        float k1 = (float)kh[base + (size_t)j * 64 + lane + 32];
        float p = q0 * k0 + q1 * k1;
#pragma unroll
        for (int off = 16; off; off >>= 1) p += __shfl_xor(p, off, 32);
        sc[t] = valid ? p : -3.0e38f;
    }

    float mx = sc[0];
#pragma unroll
    for (int t = 1; t < 9; ++t) mx = fmaxf(mx, sc[t]);
    float pr[9], sum = 0.0f;
#pragma unroll
    for (int t = 0; t < 9; ++t) { pr[t] = __expf(sc[t] - mx); sum += pr[t]; }
    float inv = 1.0f / sum;

    float o0 = 0.0f, o1 = 0.0f;
#pragma unroll
    for (int t = 0; t < 9; ++t) {
        float w = pr[t] * inv;
        o0 += w * (float)vh[base + (size_t)jn[t] * 64 + lane];
        o1 += w * (float)vh[base + (size_t)jn[t] * 64 + lane + 32];
    }

    size_t yi = ((size_t)b * NTOK + i) * DIMC + h * 64;
    yh[yi + lane]      = (f16)o0;
    yh[yi + lane + 32] = (f16)o1;
}

// ---------------------------------------------------------------------------
extern "C" void kernel_launch(void* const* d_in, const int* in_sizes, int n_in,
                              void* d_out, int out_size, void* d_ws, size_t ws_size,
                              hipStream_t stream) {
    const float* x      = (const float*)d_in[0];
    const float* w_qkv  = (const float*)d_in[1];
    const float* w_proj = (const float*)d_in[2];
    const float* b_proj = (const float*)d_in[3];
    float* out = (float*)d_out;

    const size_t XE = (size_t)BATCH * NTOK * DIMC;   // 16,777,216
    const size_t WQ = (size_t)3 * DIMC * DIMC;       //  3,145,728
    const size_t WP = (size_t)DIMC * DIMC;           //  1,048,576

    char* ws = (char*)d_ws;
    f16* xh     = (f16*)ws; ws += XE * 2;
    f16* wqkvh  = (f16*)ws; ws += WQ * 2;
    f16* wprojh = (f16*)ws; ws += WP * 2;
    f16* qh     = (f16*)ws; ws += XE * 2;
    f16* kh     = (f16*)ws; ws += XE * 2;
    f16* vh     = (f16*)ws; ws += XE * 2;
    f16* yh     = (f16*)ws; ws += XE * 2;   // total ~168 MiB

    cvt_f32_f16<<<4096, 256, 0, stream>>>(x,      xh,     XE);
    cvt_f32_f16<<<2048, 256, 0, stream>>>(w_qkv,  wqkvh,  WQ);
    cvt_f32_f16<<<1024, 256, 0, stream>>>(w_proj, wprojh, WP);

    // QKV GEMM: M=16384, N=3072, K=1024
    gemm_wmma_f16<0><<<dim3(3 * DIMC / 128, BATCH * NTOK / 128), 256, 0, stream>>>(
        xh, wqkvh, BATCH * NTOK, 3 * DIMC, DIMC, qh, kh, vh, nullptr, nullptr);

    // Local attention: B*H*N waves, 8 waves/block
    attn_local<<<(BATCH * HEADS * NTOK) / 8, 256, 0, stream>>>(qh, kh, vh, yh);

    // Proj GEMM + bias: M=16384, N=1024, K=1024 -> fp32 d_out
    gemm_wmma_f16<1><<<dim3(DIMC / 128, BATCH * NTOK / 128), 256, 0, stream>>>(
        yh, wprojh, BATCH * NTOK, DIMC, DIMC, nullptr, nullptr, nullptr, b_proj, out);
}
